// QuantizedLinearLayer_9990093930688
// MI455X (gfx1250) — compile-verified, matrix operand order checked by reading, not run on hardware
//
#include <hip/hip_runtime.h>

// LLaMA-7B FFN up-proj, AWQ w4 g128 — fused dequant + f16 WMMA GEMM for gfx1250.
// M = B*S = 8192, N = O = 11008, K = I = 4096.

typedef __attribute__((ext_vector_type(16))) _Float16 v16h;
typedef __attribute__((ext_vector_type(8)))  _Float16 v8h;
typedef __attribute__((ext_vector_type(4)))  _Float16 v4h;
typedef __attribute__((ext_vector_type(8)))  float    v8f;

#define O_DIM   11008
#define I_DIM   4096
#define GRP     128
#define M_DIM   8192   // B*S

#define BM      128
#define BN      128
#define BK      64
#define THREADS 256    // 8 wave32s

__global__ __launch_bounds__(THREADS)
void awq_w4_gemm_wmma(const _Float16* __restrict__ x,       // (M, I) fp16
                      const int*      __restrict__ packed,  // (O/4, I) int32 (int16 nibble pack)
                      const _Float16* __restrict__ scales,  // (G, O) fp16
                      const _Float16* __restrict__ zeros,   // (G, O) fp16 (= -scale*zp)
                      const _Float16* __restrict__ bias,    // (O,) fp16
                      _Float16*       __restrict__ out)     // (M, O) fp16
{
    __shared__ _Float16 As[BM * BK];   // 16 KB: activations, row-major (m, k)
    __shared__ _Float16 Bs[BN * BK];   // 16 KB: dequantized weights, (n, k) k-contiguous

    const int tid    = threadIdx.x;
    const int lane   = tid & 31;
    const int wid    = tid >> 5;       // 0..7
    const int waveM  = wid & 1;        // 2 waves along M
    const int waveN  = wid >> 1;       // 4 waves along N
    const int halfid = lane >> 4;      // ISA half-wave for fragment K-split
    const int lr     = lane & 15;

    const int n0 = blockIdx.x * BN;    // output-feature tile origin
    const int m0 = blockIdx.y * BM;    // token tile origin

    // Dequant-stage mapping: 2048 (packed-row, k) pairs per tile, 8 per thread.
    const int dq_o4 = tid >> 3;              // 0..31  packed row within tile (4 out rows each)
    const int dq_i  = (tid & 7) << 3;        // 0,8,...,56  k start (8 consecutive ints)

    const v8f vzero = {};
    v8f acc[4][2];
    #pragma unroll
    for (int i = 0; i < 4; ++i)
        #pragma unroll
        for (int j = 0; j < 2; ++j)
            acc[i][j] = vzero;

    for (int k0 = 0; k0 < I_DIM; k0 += BK) {
        __syncthreads();   // previous iteration's LDS reads complete

        // ---------------- stage A: 128x64 fp16, b128 transfers ----------------
        #pragma unroll
        for (int ss = 0; ss < 4; ++ss) {
            int seg = ss * THREADS + tid;      // 1024 16-byte segments
            int row = seg >> 3;                // 0..127
            int cs  = (seg & 7) << 3;          // 0..56 step 8 halves
            *(v8h*)&As[row * BK + cs] =
                *(const v8h*)&x[(size_t)(m0 + row) * I_DIM + k0 + cs];
        }

        // Prefetch next A tile into the cache hierarchy (global_prefetch_b8).
        if (k0 + BK < I_DIM)
            __builtin_prefetch(&x[(size_t)(m0 + (tid >> 1)) * I_DIM + k0 + BK], 0, 0);

        // ------------- stage B: fused int4 dequant -> 128x64 fp16 -------------
        {
            const int g = k0 / GRP;            // group uniform across BK=64 tile
            const size_t so = (size_t)g * O_DIM + n0 + dq_o4 * 4;
            v4h sc = *(const v4h*)&scales[so];
            v4h zp = *(const v4h*)&zeros[so];

            const int* prow = packed + (size_t)(n0 / 4 + dq_o4) * I_DIM + k0 + dq_i;
            int4 p0 = *(const int4*)prow;
            int4 p1 = *(const int4*)(prow + 4);
            int pv[8] = {p0.x, p0.y, p0.z, p0.w, p1.x, p1.y, p1.z, p1.w};

            #pragma unroll
            for (int e = 0; e < 8; ++e) {
                int p    = pv[e];
                int kcol = dq_i + e;
                #pragma unroll
                for (int j = 0; j < 4; ++j) {  // 4 output rows per packed row
                    int nib = (p >> (4 * j)) & 0xF;
                    _Float16 w = (_Float16)nib * sc[j] + zp[j];
                    Bs[(dq_o4 * 4 + j) * BK + kcol] = w;
                }
            }
        }

        __syncthreads();

        // ---------------- compute: 2 k-steps x (4x2) WMMA tiles ----------------
        #pragma unroll
        for (int kk = 0; kk < BK; kk += 32) {
            v16h afrag[4], bfrag[2];

            // 16-bit A 16x32 fragment layout: lanes 0-15 -> K {0..7,16..23},
            // lanes 16-31 -> K {8..15,24..31}; two ds_load_b128 each.
            #pragma unroll
            for (int mt = 0; mt < 4; ++mt) {
                const _Float16* ap =
                    &As[(waveM * 64 + mt * 16 + lr) * BK + kk + halfid * 8];
                union { v16h v; v8h h[2]; } u;
                u.h[0] = *(const v8h*)ap;
                u.h[1] = *(const v8h*)(ap + 16);
                afrag[mt] = u.v;
            }
            #pragma unroll
            for (int nt = 0; nt < 2; ++nt) {
                const _Float16* bp =
                    &Bs[(waveN * 32 + nt * 16 + lr) * BK + kk + halfid * 8];
                union { v16h v; v8h h[2]; } u;
                u.h[0] = *(const v8h*)bp;
                u.h[1] = *(const v8h*)(bp + 16);
                bfrag[nt] = u.v;
            }

            #pragma unroll
            for (int mt = 0; mt < 4; ++mt)
                #pragma unroll
                for (int nt = 0; nt < 2; ++nt)
                    acc[mt][nt] = __builtin_amdgcn_wmma_f32_16x16x32_f16(
                        false, afrag[mt], false, bfrag[nt],
                        (short)0, acc[mt][nt], false, false);
        }
    }

    // ---------------- epilogue: acc(f32) + bias -> fp16 ----------------
    // C/D layout: VGPR r holds M = r (+8 for lanes 16-31), N = lane & 15.
    #pragma unroll
    for (int nt = 0; nt < 2; ++nt) {
        const int n = n0 + waveN * 32 + nt * 16 + lr;
        const float bv = (float)bias[n];
        #pragma unroll
        for (int mt = 0; mt < 4; ++mt) {
            #pragma unroll
            for (int r = 0; r < 8; ++r) {
                const int m = m0 + waveM * 64 + mt * 16 + halfid * 8 + r;
                out[(size_t)m * O_DIM + n] = (_Float16)(acc[mt][nt][r] + bv);
            }
        }
    }
}

extern "C" void kernel_launch(void* const* d_in, const int* in_sizes, int n_in,
                              void* d_out, int out_size, void* d_ws, size_t ws_size,
                              hipStream_t stream) {
    (void)in_sizes; (void)n_in; (void)out_size; (void)d_ws; (void)ws_size;
    const _Float16* x    = (const _Float16*)d_in[0];
    const int*      pw   = (const int*)d_in[1];
    const _Float16* sc   = (const _Float16*)d_in[2];
    const _Float16* zp   = (const _Float16*)d_in[3];
    const _Float16* bias = (const _Float16*)d_in[4];
    _Float16*       out  = (_Float16*)d_out;

    dim3 grid(O_DIM / BN, M_DIM / BM);   // 86 x 64 workgroups
    awq_w4_gemm_wmma<<<grid, THREADS, 0, stream>>>(x, pw, sc, zp, bias, out);
}
